// CTM_app2_82815559402216
// MI455X (gfx1250) — compile-verified, matrix operand order checked by reading, not run on hardware
//
#include <hip/hip_runtime.h>
#include <hip/hip_bf16.h>
#include <math.h>

// ---------------- problem constants (match setup_inputs) ----------------
#define BB    8
#define NN    256          // tokens
#define CC    256          // input channels
#define N0    16384        // original points
#define COUT  512
#define HH    64
#define WW    64
#define HW    (HH*WW)      // 4096
#define OH    32
#define OW    32
#define OHW   (OH*OW)      // 1024
#define NS    64           // Ns = ceil(256*0.25)
#define KNN   5
#define KCONV 2304         // 9*256
#define EPSF  1e-6f
#define RSC   0.044194173824159216f   // 1/sqrt(512)

typedef __bf16 bf16_t;
typedef __attribute__((ext_vector_type(16))) __bf16 v16bf;
typedef __attribute__((ext_vector_type(8)))  __bf16 v8bf;
typedef __attribute__((ext_vector_type(8)))  float  v8f;

// ---------------- workspace layout (bytes) ----------------
static constexpr size_t OFF_SUMS   = 0;                          // 33554432 (B*HW*C f32)  [reused later]
static constexpr size_t OFF_YCONV  = 0;                          // 16777216 (B*OHW*COUT f32), reuse of SUMS
static constexpr size_t OFF_TOKS   = 16777216;                   // 4194304  (B*N*COUT f32), reuse of SUMS
static constexpr size_t OFF_TOKW   = OFF_TOKS + 4194304;         // 8192
static constexpr size_t OFF_AWD    = 0;                          // 524288 (B*N0 f32), reuse (late)
static constexpr size_t OFF_CNT    = 33554432;                   // 131072
static constexpr size_t OFF_XMAPB  = OFF_CNT + 131072;           // 16777216 bf16 channel-last map
static constexpr size_t OFF_WBT    = OFF_XMAPB + 16777216;       // 2359296  bf16 [COUT][KCONV]
static constexpr size_t OFF_XBF    = OFF_WBT + 2359296;          // 1048576  bf16 x
static constexpr size_t OFF_WSK    = OFF_XBF + 1048576;          // 262144   bf16 skip_w
static constexpr size_t OFF_XTOK   = OFF_WSK + 262144;           // 4194304  f32 x_tok / x_ln (in-place)
static constexpr size_t OFF_XLNB   = OFF_XTOK + 4194304;         // 2097152  bf16 x_ln
static constexpr size_t OFF_WEIGHT = OFF_XLNB + 2097152;         // 8192
static constexpr size_t OFF_SQ     = OFF_WEIGHT + 8192;          // 8192
static constexpr size_t OFF_DIST   = OFF_SQ + 8192;              // 2097152
static constexpr size_t OFF_DENS   = OFF_DIST + 2097152;         // 8192
static constexpr size_t OFF_SCORE  = OFF_DENS + 8192;            // 8192
static constexpr size_t OFF_DMAX   = OFF_SCORE + 8192;           // 256 (u32 bits of f32 max)
static constexpr size_t OFF_IDXD   = OFF_DMAX + 256;             // 2048 int
static constexpr size_t OFF_IDXC   = OFF_IDXD + 2048;            // 8192 int
static constexpr size_t OFF_ALLW   = OFF_IDXC + 8192;            // 2048
static constexpr size_t OFF_NORMW  = OFF_ALLW + 2048;            // 8192
static constexpr size_t OFF_AWMAX  = OFF_NORMW + 8192;           // 256 (u32 bits)
static constexpr size_t WS_NEEDED  = OFF_AWMAX + 256;            // ~62.6 MB

// ---------------- WMMA fragment helpers ----------------
// v16bf per lane = two contiguous runs of 8 bf16 (16B each):
//   lanes 0-15  : k-offsets {0..7} and {16..23}
//   lanes 16-31 : k-offsets {8..15} and {24..31}
__device__ __forceinline__ v16bf frag_load(const bf16_t* p0, const bf16_t* p1) {
  v8bf a = *(const v8bf*)p0;
  v8bf b = *(const v8bf*)p1;
  v16bf r;
#pragma unroll
  for (int i = 0; i < 8; ++i) { r[i] = a[i]; r[i + 8] = b[i]; }
  return r;
}
__device__ __forceinline__ v16bf frag_zero() {
  v16bf r;
#pragma unroll
  for (int i = 0; i < 16; ++i) r[i] = (bf16_t)0.0f;
  return r;
}

// ---------------- prep kernels ----------------
__global__ __launch_bounds__(256) void k_cvt_x_bf16(const float* __restrict__ x, bf16_t* __restrict__ o) {
  size_t g = (size_t)blockIdx.x * 256 + threadIdx.x;   // B*N*C = 524288
  o[g] = (bf16_t)x[g];
}
__global__ __launch_bounds__(256) void k_cvt_skip_bf16(const float* __restrict__ w, bf16_t* __restrict__ o) {
  size_t g = (size_t)blockIdx.x * 256 + threadIdx.x;   // COUT*CC = 131072
  o[g] = (bf16_t)w[g];
}
// conv_w (COUT,CC,3,3) -> wBt[o][ (ky*3+kx)*CC + c ]
__global__ __launch_bounds__(256) void k_build_wbt(const float* __restrict__ cw, bf16_t* __restrict__ wbt) {
  size_t g = (size_t)blockIdx.x * 256 + threadIdx.x;   // COUT*KCONV = 1179648
  int o = (int)(g / KCONV);
  int k = (int)(g - (size_t)o * KCONV);
  int c = k & 255;
  int p = k >> 8;                  // 0..8
  int ky = p / 3, kx = p - ky * 3;
  wbt[g] = (bf16_t)cw[(((size_t)o * CC + c) * 3 + ky) * 3 + kx];
}

// ---------------- token2map scatter ----------------
__device__ __forceinline__ int cell_of(float lx, float ly, int Hc, int Wc) {
  lx = fminf(fmaxf(lx, -1.0f), 1.0f);
  ly = fminf(fmaxf(ly, -1.0f), 1.0f);
  int px = (int)roundf(0.5f * (lx + 1.0f) * (float)Wc - 0.5f);
  int py = (int)roundf(0.5f * (ly + 1.0f) * (float)Hc - 0.5f);
  px = min(max(px, 0), Wc - 1);
  py = min(max(py, 0), Hc - 1);
  return py * Wc + px;
}

__global__ __launch_bounds__(256) void k_scatter_map(const float* __restrict__ x,
    const float* __restrict__ loc, const int* __restrict__ idx_agg,
    float* __restrict__ sums, float* __restrict__ cnt) {
  int blk = blockIdx.x;                // B*N0 blocks
  int b = blk >> 14, n0 = blk & (N0 - 1);
  size_t g = (size_t)b * N0 + n0;
  int cell = cell_of(loc[g * 2 + 0], loc[g * 2 + 1], HH, WW);
  int seg  = b * HW + cell;
  int tok  = idx_agg[g];
  float f = x[((size_t)b * NN + tok) * CC + threadIdx.x];
  atomicAdd(&sums[(size_t)seg * CC + threadIdx.x], f);
  if (threadIdx.x == 0) atomicAdd(&cnt[seg], 1.0f);
}

__global__ __launch_bounds__(256) void k_finalize_map(const float* __restrict__ sums,
    const float* __restrict__ cnt, bf16_t* __restrict__ xmapb) {
  int seg = blockIdx.x;                // B*HW
  float inv = 1.0f / (cnt[seg] + EPSF);
  size_t base = (size_t)seg * CC + threadIdx.x;
  xmapb[base] = (bf16_t)(sums[base] * inv);
}

// ---------------- conv as implicit GEMM (WMMA bf16) ----------------
// grid (8 Mtiles, 8 Ntiles, B), 256 threads = 8 waves, wave strip 16x64
__global__ __launch_bounds__(256) void k_conv_wmma(const bf16_t* __restrict__ xmap,
    const bf16_t* __restrict__ wbt, const float* __restrict__ convb,
    float* __restrict__ yconv) {
  const int wave = threadIdx.x >> 5, lane = threadIdx.x & 31;
  const int lhalf = lane >> 4, l16 = lane & 15;
  const int b = blockIdx.z;
  const int m_base = blockIdx.x * 128 + wave * 16;
  const int n_base = blockIdx.y * 64;
  const int mrow = m_base + l16;
  const int oy = mrow >> 5, ox = mrow & 31;
  const int kofs = lhalf * 8;

  v8f acc[4];
#pragma unroll
  for (int t = 0; t < 4; ++t)
#pragma unroll
    for (int j = 0; j < 8; ++j) acc[t][j] = 0.0f;

  for (int kk = 0; kk < KCONV; kk += 32) {
    int ky = kk / 768;
    int r  = kk - ky * 768;
    int kx = r >> 8;
    int c0 = (r & 255) + kofs;
    int iy = 2 * oy - 1 + ky;
    int ix = 2 * ox - 1 + kx;
    v16bf a;
    if (iy >= 0 && iy < HH && ix >= 0 && ix < WW) {
      const bf16_t* ap = xmap + (((size_t)b * HW + iy * WW + ix) * CC + c0);
      __builtin_prefetch(ap + 32, 0, 1);     // global_prefetch_b8
      a = frag_load(ap, ap + 16);
    } else {
      a = frag_zero();
    }
#pragma unroll
    for (int t = 0; t < 4; ++t) {
      const bf16_t* bp = wbt + ((size_t)(n_base + t * 16 + l16) * KCONV + kk + kofs);
      v16bf bf = frag_load(bp, bp + 16);
      acc[t] = __builtin_amdgcn_wmma_f32_16x16x32_bf16(false, a, false, bf,
                                                       (short)0, acc[t], false, false);
    }
  }
#pragma unroll
  for (int t = 0; t < 4; ++t) {
    int n = n_base + t * 16 + l16;
    float bias = convb[n];
#pragma unroll
    for (int j = 0; j < 8; ++j) {
      int m = m_base + j + (lhalf << 3);
      yconv[((size_t)b * OHW + m) * COUT + n] = acc[t][j] + bias;
    }
  }
}

// ---------------- map2token gather ----------------
__global__ __launch_bounds__(256) void k_gather_tokens(const float* __restrict__ yconv,
    const float* __restrict__ loc, const int* __restrict__ idx_agg,
    const float* __restrict__ aggw, float* __restrict__ toks, float* __restrict__ tokw) {
  int blk = blockIdx.x;                // B*N0
  int b = blk >> 14, n0 = blk & (N0 - 1);
  size_t g = (size_t)b * N0 + n0;
  int cell = cell_of(loc[g * 2 + 0], loc[g * 2 + 1], OH, OW);
  int tok = idx_agg[g];
  float w = aggw[g];
  size_t src = ((size_t)b * OHW + cell) * COUT;
  size_t dst = ((size_t)b * NN + tok) * COUT;
  int c = threadIdx.x;
  atomicAdd(&toks[dst + c], w * yconv[src + c]);
  atomicAdd(&toks[dst + c + 256], w * yconv[src + c + 256]);
  if (c == 0) atomicAdd(&tokw[b * NN + tok], w);
}

// ---------------- skip GEMM + map2token combine (WMMA bf16) ----------------
// grid (16, 8), M=B*N=2048, N=COUT, K=CC
__global__ __launch_bounds__(256) void k_skip_wmma(const bf16_t* __restrict__ xbf,
    const bf16_t* __restrict__ wsk, const float* __restrict__ toks,
    const float* __restrict__ tokw, float* __restrict__ xtok) {
  const int wave = threadIdx.x >> 5, lane = threadIdx.x & 31;
  const int lhalf = lane >> 4, l16 = lane & 15;
  const int m_base = blockIdx.x * 128 + wave * 16;
  const int n_base = blockIdx.y * 64;
  const int kofs = lhalf * 8;

  v8f acc[4];
#pragma unroll
  for (int t = 0; t < 4; ++t)
#pragma unroll
    for (int j = 0; j < 8; ++j) acc[t][j] = 0.0f;

  const bf16_t* arow = xbf + (size_t)(m_base + l16) * CC;
  for (int kk = 0; kk < CC; kk += 32) {
    v16bf a = frag_load(arow + kk + kofs, arow + kk + kofs + 16);
#pragma unroll
    for (int t = 0; t < 4; ++t) {
      const bf16_t* bp = wsk + ((size_t)(n_base + t * 16 + l16) * CC + kk + kofs);
      v16bf bf = frag_load(bp, bp + 16);
      acc[t] = __builtin_amdgcn_wmma_f32_16x16x32_bf16(false, a, false, bf,
                                                       (short)0, acc[t], false, false);
    }
  }
#pragma unroll
  for (int t = 0; t < 4; ++t) {
    int n = n_base + t * 16 + l16;
#pragma unroll
    for (int j = 0; j < 8; ++j) {
      int m = m_base + j + (lhalf << 3);
      float mt = toks[(size_t)m * COUT + n] / (tokw[m] + EPSF);
      xtok[(size_t)m * COUT + n] = mt + acc[t][j];
    }
  }
}

// ---------------- LayerNorm + conf + exp ----------------
__global__ __launch_bounds__(256) void k_layernorm(float* __restrict__ xtok,
    bf16_t* __restrict__ xlnb, const float* __restrict__ lng, const float* __restrict__ lnb,
    const float* __restrict__ cw, const float* __restrict__ cb,
    float* __restrict__ sq, float* __restrict__ wgt) {
  int m = blockIdx.x, t = threadIdx.x;           // 2048 rows
  size_t base = (size_t)m * COUT;
  float v0 = xtok[base + t], v1 = xtok[base + t + 256];
  __shared__ float s1[256], s2[256];
  s1[t] = v0 + v1;
  s2[t] = v0 * v0 + v1 * v1;
  __syncthreads();
  for (int off = 128; off > 0; off >>= 1) {
    if (t < off) { s1[t] += s1[t + off]; s2[t] += s2[t + off]; }
    __syncthreads();
  }
  float mu  = s1[0] * (1.0f / COUT);
  float var = fmaxf(s2[0] * (1.0f / COUT) - mu * mu, 0.0f);
  float rs  = rsqrtf(var + 1e-5f);
  __syncthreads();
  float y0 = (v0 - mu) * rs * lng[t] + lnb[t];
  float y1 = (v1 - mu) * rs * lng[t + 256] + lnb[t + 256];
  xtok[base + t] = y0;  xtok[base + t + 256] = y1;
  xlnb[base + t] = (bf16_t)y0;  xlnb[base + t + 256] = (bf16_t)y1;
  s1[t] = y0 * y0 + y1 * y1;
  s2[t] = y0 * cw[t] + y1 * cw[t + 256];
  __syncthreads();
  for (int off = 128; off > 0; off >>= 1) {
    if (t < off) { s1[t] += s1[t + off]; s2[t] += s2[t + off]; }
    __syncthreads();
  }
  if (t == 0) { sq[m] = s1[0]; wgt[m] = expf(s2[0] + cb[0]); }
}

// ---------------- Gram GEMM -> pairwise distances (WMMA bf16) ----------------
// grid (2, 4, B), M=N=256, K=512
__global__ __launch_bounds__(256) void k_gram_wmma(const bf16_t* __restrict__ xlnb,
    const float* __restrict__ sq, float* __restrict__ dist, unsigned* __restrict__ dmax) {
  const int wave = threadIdx.x >> 5, lane = threadIdx.x & 31;
  const int lhalf = lane >> 4, l16 = lane & 15;
  const int b = blockIdx.z;
  const int m_base = blockIdx.x * 128 + wave * 16;
  const int n_base = blockIdx.y * 64;
  const int kofs = lhalf * 8;
  const bf16_t* xb = xlnb + (size_t)b * NN * COUT;

  v8f acc[4];
#pragma unroll
  for (int t = 0; t < 4; ++t)
#pragma unroll
    for (int j = 0; j < 8; ++j) acc[t][j] = 0.0f;

  const bf16_t* arow = xb + (size_t)(m_base + l16) * COUT;
  for (int kk = 0; kk < COUT; kk += 32) {
    v16bf a = frag_load(arow + kk + kofs, arow + kk + kofs + 16);
#pragma unroll
    for (int t = 0; t < 4; ++t) {
      const bf16_t* bp = xb + (size_t)(n_base + t * 16 + l16) * COUT + kk + kofs;
      v16bf bf = frag_load(bp, bp + 16);
      acc[t] = __builtin_amdgcn_wmma_f32_16x16x32_bf16(false, a, false, bf,
                                                       (short)0, acc[t], false, false);
    }
  }
  float lmax = 0.0f;
#pragma unroll
  for (int t = 0; t < 4; ++t) {
    int jc = n_base + t * 16 + l16;
    float sj = sq[b * NN + jc];
#pragma unroll
    for (int j = 0; j < 8; ++j) {
      int i = m_base + j + (lhalf << 3);
      float d2 = fmaxf(sq[b * NN + i] + sj - 2.0f * acc[t][j], 0.0f);
      float d = sqrtf(d2) * RSC;
      dist[((size_t)b * NN + i) * NN + jc] = d;
      lmax = fmaxf(lmax, d);
    }
  }
  atomicMax(&dmax[b], __float_as_uint(lmax));
}

// ---------------- density (top-5 nearest) ----------------
__global__ __launch_bounds__(256) void k_density(const float* __restrict__ dist,
    float* __restrict__ dens) {
  int row = blockIdx.x, t = threadIdx.x;        // row = b*NN + i
  float d = dist[(size_t)row * NN + t];
  __shared__ float sv[256];
  __shared__ int   si[256];
  float accum = 0.0f;
  for (int r = 0; r < KNN; ++r) {
    sv[t] = d; si[t] = t;
    __syncthreads();
    for (int off = 128; off > 0; off >>= 1) {
      if (t < off) {
        if (sv[t + off] < sv[t] || (sv[t + off] == sv[t] && si[t + off] < si[t])) {
          sv[t] = sv[t + off]; si[t] = si[t + off];
        }
      }
      __syncthreads();
    }
    float mv = sv[0]; int mi = si[0];
    accum += mv * mv;
    __syncthreads();
    if (t == mi) d = 3.4e38f;
  }
  if (t == 0) dens[row] = expf(-accum * (1.0f / KNN));
}

// ---------------- parent distance + score ----------------
__global__ __launch_bounds__(256) void k_score(const float* __restrict__ dist,
    const float* __restrict__ dens, const unsigned* __restrict__ dmax,
    float* __restrict__ score) {
  int row = blockIdx.x, t = threadIdx.x;        // row = b*NN + i
  int b = row >> 8;
  float di = dens[row];
  float dmaxb = __uint_as_float(dmax[b]);
  float v = (dens[b * NN + t] > di) ? dist[(size_t)row * NN + t] : dmaxb;
  __shared__ float sv[256];
  sv[t] = v;
  __syncthreads();
  for (int off = 128; off > 0; off >>= 1) {
    if (t < off) sv[t] = fminf(sv[t], sv[t + off]);
    __syncthreads();
  }
  if (t == 0) score[row] = sv[0] * di;
}

// ---------------- top-Ns scores per batch (descending) ----------------
__global__ __launch_bounds__(256) void k_topns(const float* __restrict__ score,
    int* __restrict__ idxd) {
  int b = blockIdx.x, t = threadIdx.x;
  __shared__ float vals[256];
  __shared__ float sv[256];
  __shared__ int   si[256];
  vals[t] = score[b * NN + t];
  __syncthreads();
  for (int s = 0; s < NS; ++s) {
    sv[t] = vals[t]; si[t] = t;
    __syncthreads();
    for (int off = 128; off > 0; off >>= 1) {
      if (t < off) {
        if (sv[t + off] > sv[t] || (sv[t + off] == sv[t] && si[t + off] < si[t])) {
          sv[t] = sv[t + off]; si[t] = si[t + off];
        }
      }
      __syncthreads();
    }
    int win = si[0];
    if (t == 0) idxd[b * NS + s] = win;
    __syncthreads();
    if (t == win) vals[t] = -3.4e38f;
    __syncthreads();
  }
}

// ---------------- cluster assignment ----------------
__global__ __launch_bounds__(64) void k_assign(const float* __restrict__ dist,
    const int* __restrict__ idxd, int* __restrict__ idxc) {
  int row = blockIdx.x, t = threadIdx.x;        // row = b*NN + i, t = candidate s
  int b = row >> 8, i = row & 255;
  __shared__ float sv[64];
  __shared__ int   si[64];
  int j = idxd[b * NS + t];
  sv[t] = dist[((size_t)b * NN + j) * NN + i];
  si[t] = t;
  __syncthreads();
  for (int off = 32; off > 0; off >>= 1) {
    if (t < off) {
      if (sv[t + off] < sv[t] || (sv[t + off] == sv[t] && si[t + off] < si[t])) {
        sv[t] = sv[t + off]; si[t] = si[t + off];
      }
    }
    __syncthreads();
  }
  if (t == 0) idxc[row] = si[0];
}

__global__ __launch_bounds__(64) void k_seed(const int* __restrict__ idxd,
    int* __restrict__ idxc) {
  int b = blockIdx.x, s = threadIdx.x;
  idxc[b * NN + idxd[b * NS + s]] = s;
}

// ---------------- segment sums for merge ----------------
__global__ __launch_bounds__(256) void k_allw(const int* __restrict__ idxc,
    const float* __restrict__ wgt, float* __restrict__ allw) {
  int b = blockIdx.x, t = threadIdx.x;
  atomicAdd(&allw[b * NS + idxc[b * NN + t]], wgt[b * NN + t]);
}

__global__ __launch_bounds__(256) void k_normw(const int* __restrict__ idxc,
    const float* __restrict__ wgt, const float* __restrict__ allw,
    float* __restrict__ normw) {
  int b = blockIdx.x, t = threadIdx.x;
  int g = b * NN + t;
  normw[g] = wgt[g] / (allw[b * NS + idxc[g]] + EPSF);
}

__global__ __launch_bounds__(256) void k_merge(const float* __restrict__ xln,
    const int* __restrict__ idxc, const float* __restrict__ normw,
    float* __restrict__ xout) {
  int row = blockIdx.x;                         // b*NN + i
  int b = row >> 8;
  int cl = idxc[row];
  float nw = normw[row];
  size_t src = (size_t)row * COUT;
  size_t dst = ((size_t)b * NS + cl) * COUT;
  int c = threadIdx.x;
  atomicAdd(&xout[dst + c],        xln[src + c] * nw);
  atomicAdd(&xout[dst + c + 256],  xln[src + c + 256] * nw);
}

// ---------------- downstream index / weight outputs ----------------
__global__ __launch_bounds__(256) void k_downidx(const int* __restrict__ idx_agg,
    const int* __restrict__ idxc, const float* __restrict__ normw,
    const float* __restrict__ aggw, float* __restrict__ out_idx,
    float* __restrict__ awd, unsigned* __restrict__ awmax) {
  size_t g = (size_t)blockIdx.x * 256 + threadIdx.x;   // B*N0
  int b = (int)(g >> 14);
  int tok = idx_agg[g];
  int cl = idxc[b * NN + tok];
  float nw = normw[b * NN + tok];
  float v = aggw[g] * nw;
  out_idx[g] = (float)cl;
  awd[g] = v;
  atomicMax(&awmax[b], __float_as_uint(v));
}

__global__ __launch_bounds__(256) void k_downw(const float* __restrict__ awd,
    const unsigned* __restrict__ awmax, float* __restrict__ out_aw) {
  size_t g = (size_t)blockIdx.x * 256 + threadIdx.x;
  int b = (int)(g >> 14);
  out_aw[g] = awd[g] / __uint_as_float(awmax[b]);
}

// ---------------- launch ----------------
extern "C" void kernel_launch(void* const* d_in, const int* in_sizes, int n_in,
                              void* d_out, int out_size, void* d_ws, size_t ws_size,
                              hipStream_t stream) {
  if (ws_size < WS_NEEDED) return;

  const float* x        = (const float*)d_in[0];
  const float* loc      = (const float*)d_in[1];
  const int*   idx_agg  = (const int*)  d_in[2];
  const float* aggw     = (const float*)d_in[3];
  // d_in[4] = idx_k_loc (unused by reference forward), d_in[5]/[6] = H/W (fixed 64)
  const float* conv_w   = (const float*)d_in[7];
  const float* conv_b   = (const float*)d_in[8];
  const float* skip_w   = (const float*)d_in[9];
  const float* ln_g     = (const float*)d_in[10];
  const float* ln_b     = (const float*)d_in[11];
  const float* conf_w   = (const float*)d_in[12];
  const float* conf_b   = (const float*)d_in[13];

  char* ws = (char*)d_ws;
  float*    sums   = (float*)   (ws + OFF_SUMS);
  float*    yconv  = (float*)   (ws + OFF_YCONV);
  float*    toks   = (float*)   (ws + OFF_TOKS);
  float*    tokw   = (float*)   (ws + OFF_TOKW);
  float*    awd    = (float*)   (ws + OFF_AWD);
  float*    cnt    = (float*)   (ws + OFF_CNT);
  bf16_t*   xmapb  = (bf16_t*)  (ws + OFF_XMAPB);
  bf16_t*   wbt    = (bf16_t*)  (ws + OFF_WBT);
  bf16_t*   xbf    = (bf16_t*)  (ws + OFF_XBF);
  bf16_t*   wsk    = (bf16_t*)  (ws + OFF_WSK);
  float*    xtok   = (float*)   (ws + OFF_XTOK);
  bf16_t*   xlnb   = (bf16_t*)  (ws + OFF_XLNB);
  float*    wgt    = (float*)   (ws + OFF_WEIGHT);
  float*    sq     = (float*)   (ws + OFF_SQ);
  float*    dist   = (float*)   (ws + OFF_DIST);
  float*    dens   = (float*)   (ws + OFF_DENS);
  float*    score  = (float*)   (ws + OFF_SCORE);
  unsigned* dmax   = (unsigned*)(ws + OFF_DMAX);
  int*      idxd   = (int*)     (ws + OFF_IDXD);
  int*      idxc   = (int*)     (ws + OFF_IDXC);
  float*    allw   = (float*)   (ws + OFF_ALLW);
  float*    normw  = (float*)   (ws + OFF_NORMW);
  unsigned* awmax  = (unsigned*)(ws + OFF_AWMAX);

  float* out_x   = (float*)d_out;                       // B*NS*COUT = 262144
  float* out_idx = out_x + (size_t)BB * NS * COUT;      // B*N0      = 131072
  float* out_aw  = out_idx + (size_t)BB * N0;           // B*N0      = 131072

  // --- phase 0: prep / zero
  hipMemsetAsync(ws + OFF_SUMS, 0, (size_t)33554432 + 131072, stream);   // sums + cnt
  k_cvt_x_bf16   <<<2048, 256, 0, stream>>>(x, xbf);
  k_cvt_skip_bf16<<<512,  256, 0, stream>>>(skip_w, wsk);
  k_build_wbt    <<<4608, 256, 0, stream>>>(conv_w, wbt);

  // --- token2map
  k_scatter_map  <<<BB * N0, 256, 0, stream>>>(x, loc, idx_agg, sums, cnt);
  k_finalize_map <<<BB * HW, 256, 0, stream>>>(sums, cnt, xmapb);

  // sums region now free -> reuse for yconv / token sums
  hipMemsetAsync(ws + OFF_TOKS, 0, (size_t)4194304 + 8192, stream);      // toks + tokw

  // --- conv 3x3 s2 via WMMA implicit GEMM
  k_conv_wmma<<<dim3(8, 8, BB), 256, 0, stream>>>(xmapb, wbt, conv_b, yconv);

  // --- map2token gather + skip GEMM combine
  k_gather_tokens<<<BB * N0, 256, 0, stream>>>(yconv, loc, idx_agg, aggw, toks, tokw);
  k_skip_wmma<<<dim3(16, 8), 256, 0, stream>>>(xbf, wsk, toks, tokw, xtok);

  // --- layernorm + conf
  k_layernorm<<<BB * NN, 256, 0, stream>>>(xtok, xlnb, ln_g, ln_b, conf_w, conf_b, sq, wgt);

  // --- token_cluster_merge
  hipMemsetAsync(ws + OFF_DMAX,  0, 256,  stream);
  hipMemsetAsync(ws + OFF_ALLW,  0, 2048, stream);
  hipMemsetAsync(ws + OFF_AWMAX, 0, 256,  stream);
  hipMemsetAsync(d_out, 0, (size_t)BB * NS * COUT * sizeof(float), stream);

  k_gram_wmma<<<dim3(2, 4, BB), 256, 0, stream>>>(xlnb, sq, dist, dmax);
  k_density  <<<BB * NN, 256, 0, stream>>>(dist, dens);
  k_score    <<<BB * NN, 256, 0, stream>>>(dist, dens, dmax, score);
  k_topns    <<<BB,      256, 0, stream>>>(score, idxd);
  k_assign   <<<BB * NN, 64,  0, stream>>>(dist, idxd, idxc);
  k_seed     <<<BB,      64,  0, stream>>>(idxd, idxc);
  k_allw     <<<BB,      256, 0, stream>>>(idxc, wgt, allw);
  k_normw    <<<BB,      256, 0, stream>>>(idxc, wgt, allw, normw);
  k_merge    <<<BB * NN, 256, 0, stream>>>(xtok, idxc, normw, out_x);

  // --- downstream idx/weight outputs (yconv region free -> awd reuse)
  k_downidx<<<BB * N0 / 256, 256, 0, stream>>>(idx_agg, idxc, normw, aggw, out_idx, awd, awmax);
  k_downw  <<<BB * N0 / 256, 256, 0, stream>>>(awd, awmax, out_aw);
}
